// AttentionModule_28321014350504
// MI455X (gfx1250) — matrix operand order repaired and checked by reading, not compiled, not run on hardware
//
#include <hip/hip_runtime.h>
#include <math.h>

typedef _Float16 v8h  __attribute__((ext_vector_type(8)));
typedef _Float16 v16h __attribute__((ext_vector_type(16)));
typedef float    v8f  __attribute__((ext_vector_type(8)));

static __device__ __forceinline__ v8f wmma16(v16h a, v16h b, v8f c) {
  // D = A(16x32 f16) * B(32x16 f16) + C(16x16 f32)
  return __builtin_amdgcn_wmma_f32_16x16x32_f16(false, a, false, b, (short)0, c, false, false);
}

static __device__ __forceinline__ v16h cat8(v8h lo, v8h hi) {
  return __builtin_shufflevector(lo, hi, 0,1,2,3,4,5,6,7,8,9,10,11,12,13,14,15);
}

// async copy of one 16-byte chunk: global -> LDS (ASYNCcnt-tracked)
static __device__ __forceinline__ void async_b128(const _Float16* g, const _Float16* l) {
  unsigned lo = (unsigned)(unsigned long long)(const void*)l;  // low 32b of flat = LDS offset
  asm volatile("global_load_async_to_lds_b128 %0, %1, off" :: "v"(lo), "v"(g) : "memory");
}

// ---------------- weight convert f32[K,N] -> f16[N,K] (transposed) ----------------
__global__ __launch_bounds__(256) void k_wconv(const float* __restrict__ src,
                                               _Float16* __restrict__ dst, int K, int N) {
  int idx = blockIdx.x * 256 + threadIdx.x;
  if (idx < K * N) {
    int k = idx / N, n = idx % N;
    dst[(size_t)n * K + k] = (_Float16)src[idx];
  }
}

// ---------------- x + pos_embed -> f16 ----------------
__global__ __launch_bounds__(256) void k_posembed(const float* __restrict__ x,
                                                  const float* __restrict__ pos,
                                                  _Float16* __restrict__ xpe16,
                                                  int total, int SeH) {
  int idx = blockIdx.x * 256 + threadIdx.x;
  if (idx < total) xpe16[idx] = (_Float16)(x[idx] + pos[idx % SeH]);
}

// ---------------- h32 = y ; h16 = f16(y) ----------------
__global__ __launch_bounds__(256) void k_inith(const float* __restrict__ y,
                                               float* __restrict__ h32,
                                               _Float16* __restrict__ h16, int total) {
  int idx = blockIdx.x * 256 + threadIdx.x;
  if (idx < total) { float v = y[idx]; h32[idx] = v; h16[idx] = (_Float16)v; }
}

// ---------------- WMMA GEMM: C[M,N] = A[M,K](f16 rowmajor) * Wt[N,K](f16) + bias ----------------
// flags: 1=write f32 out32, 2=write f16 out16, 4=transposed f16 store (per batch of SB rows), 8=gelu
__global__ __launch_bounds__(256) void k_gemm(const _Float16* __restrict__ A,
                                              const _Float16* __restrict__ Wt,
                                              const float* __restrict__ bias,
                                              float* __restrict__ out32,
                                              _Float16* __restrict__ out16,
                                              int M, int N, int K, int flags, int SB) {
  int tid = threadIdx.x;
  int wave = tid >> 5, lane = tid & 31, hl = lane >> 4, lcol = lane & 15;
  int m0 = blockIdx.x * 128 + wave * 16;
  int n0 = blockIdx.y * 64;
  (void)M;
  v8f acc[4] = {};
  const _Float16* arow = A + (size_t)(m0 + lcol) * K;
  for (int k0 = 0; k0 < K; k0 += 32) {
    v8h alo = *(const v8h*)(arow + k0 + hl * 8);
    v8h ahi = *(const v8h*)(arow + k0 + 16 + hl * 8);
    v16h a = cat8(alo, ahi);
#pragma unroll
    for (int t = 0; t < 4; ++t) {
      const _Float16* brow = Wt + (size_t)(n0 + t * 16 + lcol) * K + k0;
      v16h b = cat8(*(const v8h*)(brow + hl * 8), *(const v8h*)(brow + 16 + hl * 8));
      acc[t] = wmma16(a, b, acc[t]);
    }
  }
#pragma unroll
  for (int t = 0; t < 4; ++t) {
    int n = n0 + t * 16 + lcol;
    float bv = bias[n];
#pragma unroll
    for (int vr = 0; vr < 8; ++vr) {
      int m = m0 + vr + 8 * hl;
      float v = acc[t][vr] + bv;
      if (flags & 8) {  // gelu (tanh approximation, JAX default)
        float u = v;
        v = 0.5f * u * (1.0f + tanhf(0.7978845608028654f * (u + 0.044715f * u * u * u)));
      }
      if (flags & 1) out32[(size_t)m * N + n] = v;
      if (flags & 2) {
        if (flags & 4) {
          int bb = m / SB, mm = m - bb * SB;
          out16[((size_t)bb * N + n) * SB + mm] = (_Float16)v;  // [B][N][SB]
        } else {
          out16[(size_t)m * N + n] = (_Float16)v;
        }
      }
    }
  }
}

// ---------------- flash attention with async double-buffered K/V staging ----------------
// Q: [B*Sq, 256] f16, K: [B*Sk, 256] f16, Vt: [B][256][Sk] f16, O: [B*Sq, 256] f16
__global__ __launch_bounds__(128) void k_flash(const _Float16* __restrict__ Q,
                                               const _Float16* __restrict__ Kb,
                                               const _Float16* __restrict__ Vt,
                                               _Float16* __restrict__ O, int Sq, int Sk) {
  const int H = 256, HD = 64;
  __shared__ alignas(64) _Float16 ktile[2][64 * 64];  // [sk 64][hd 64]
  __shared__ alignas(64) _Float16 vtile[2][64 * 64];  // [hd 64][sk 64]
  __shared__ alignas(64) _Float16 psh[4][16 * 64];
  int tid = threadIdx.x;
  int wave = tid >> 5, lane = tid & 31, hl = lane >> 4, lcol = lane & 15;
  int q0 = blockIdx.x * 64 + wave * 16;
  int head = blockIdx.y, b = blockIdx.z;

  const _Float16* kbase = Kb + (size_t)b * Sk * H + head * HD;   // + row*H
  const _Float16* vbase = Vt + ((size_t)b * H + head * HD) * Sk; // + hd*Sk

  // issue one chunk's async copies: 8 instructions/thread, j<4 -> K tile, j>=4 -> V tile
  auto issue_chunk = [&](int sk0, int buf) {
#pragma unroll
    for (int j = 0; j < 8; ++j) {
      int c = tid + 128 * j;
      if (j < 4) {
        int row = c >> 3, part = c & 7;
        async_b128(kbase + (size_t)(sk0 + row) * H + part * 8,
                   &ktile[buf][row * 64 + part * 8]);
      } else {
        int c2 = c - 512;
        int row = c2 >> 3, part = c2 & 7;
        async_b128(vbase + (size_t)row * Sk + sk0 + part * 8,
                   &vtile[buf][row * 64 + part * 8]);
      }
    }
  };

  const _Float16* qrow = Q + (size_t)(b * Sq + q0 + lcol) * H + head * HD;
  v16h qa[2];
#pragma unroll
  for (int ks = 0; ks < 2; ++ks) {
    v8h lo = *(const v8h*)(qrow + ks * 32 + hl * 8);
    v8h hi = *(const v8h*)(qrow + ks * 32 + 16 + hl * 8);
    v16h a = cat8(lo, hi);
#pragma unroll
    for (int i = 0; i < 16; ++i) a[i] = a[i] * (_Float16)0.125f;  // 1/sqrt(64), exact
    qa[ks] = a;
  }

  issue_chunk(0, 0);

  v8f o[4] = {};
  float mrow[8], lrow[8];
#pragma unroll
  for (int r = 0; r < 8; ++r) { mrow[r] = -3.0e38f; lrow[r] = 0.0f; }

  int nchunk = Sk / 64;
  for (int ci = 0; ci < nchunk; ++ci) {
    int sk0 = ci * 64;
    if (ci + 1 < nchunk) {
      issue_chunk(sk0 + 64, (ci + 1) & 1);
      asm volatile("s_wait_asynccnt 0x8" ::: "memory");  // chunk ci's 8 ops retired
    } else {
      asm volatile("s_wait_asynccnt 0x0" ::: "memory");
    }
    __syncthreads();  // all waves' copies for chunk ci visible
    const _Float16* kptr = ktile[ci & 1];
    const _Float16* vptr = vtile[ci & 1];

    v8f s[4] = {};
#pragma unroll
    for (int ks = 0; ks < 2; ++ks) {
#pragma unroll
      for (int t = 0; t < 4; ++t) {
        const _Float16* kr = kptr + (size_t)(t * 16 + lcol) * 64 + ks * 32;
        v16h bk = cat8(*(const v8h*)(kr + hl * 8), *(const v8h*)(kr + 16 + hl * 8));
        s[t] = wmma16(qa[ks], bk, s[t]);
      }
    }
    // online softmax, per output row (row = vr + 8*hl lives in one 16-lane half)
#pragma unroll
    for (int vr = 0; vr < 8; ++vr) {
      float vm = fmaxf(fmaxf(s[0][vr], s[1][vr]), fmaxf(s[2][vr], s[3][vr]));
#pragma unroll
      for (int off = 8; off >= 1; off >>= 1) vm = fmaxf(vm, __shfl_xor(vm, off));
      float mnew = fmaxf(mrow[vr], vm);
      float alpha = __expf(mrow[vr] - mnew);
      mrow[vr] = mnew;
      float rs = 0.0f;
#pragma unroll
      for (int t = 0; t < 4; ++t) {
        float p = __expf(s[t][vr] - mnew);
        s[t][vr] = p;
        rs += p;
      }
#pragma unroll
      for (int off = 8; off >= 1; off >>= 1) rs += __shfl_xor(rs, off);
      lrow[vr] = lrow[vr] * alpha + rs;
#pragma unroll
      for (int t = 0; t < 4; ++t) o[t][vr] *= alpha;
      int row = vr + 8 * hl;
#pragma unroll
      for (int t = 0; t < 4; ++t)
        psh[wave][row * 64 + t * 16 + lcol] = (_Float16)s[t][vr];
    }
    asm volatile("s_wait_dscnt 0x0" ::: "memory");  // wave-local LDS RAW fence
#pragma unroll
    for (int ks = 0; ks < 2; ++ks) {
      v8h lo = *(const v8h*)&psh[wave][lcol * 64 + ks * 32 + hl * 8];
      v8h hi = *(const v8h*)&psh[wave][lcol * 64 + ks * 32 + 16 + hl * 8];
      v16h ap = cat8(lo, hi);
#pragma unroll
      for (int t = 0; t < 4; ++t) {
        const _Float16* vr2 = vptr + (size_t)(t * 16 + lcol) * 64 + ks * 32;
        v16h bv = cat8(*(const v8h*)(vr2 + hl * 8), *(const v8h*)(vr2 + 16 + hl * 8));
        o[t] = wmma16(ap, bv, o[t]);
      }
    }
    __syncthreads();  // done reading buf[ci&1] before it is re-filled at ci+2
  }
#pragma unroll
  for (int vr = 0; vr < 8; ++vr) {
    float inv = 1.0f / lrow[vr];
    int m = b * Sq + q0 + vr + 8 * hl;
#pragma unroll
    for (int t = 0; t < 4; ++t)
      O[(size_t)m * H + head * HD + t * 16 + lcol] = (_Float16)(o[t][vr] * inv);
  }
}

// ---------------- residual + layernorm ----------------
__global__ __launch_bounds__(256) void k_ln(const float* __restrict__ base,
                                            const float* __restrict__ add,
                                            const float* __restrict__ g,
                                            const float* __restrict__ bta,
                                            float* __restrict__ h32,
                                            _Float16* __restrict__ h16,
                                            float* __restrict__ fin, int H) {
  __shared__ float s1[8], s2[8];
  int i = threadIdx.x, row = blockIdx.x;
  size_t idx = (size_t)row * H + i;
  float v = base[idx] + add[idx];
  float s = v, ss = v * v;
#pragma unroll
  for (int off = 16; off >= 1; off >>= 1) {
    s += __shfl_down(s, off);
    ss += __shfl_down(ss, off);
  }
  int wave = threadIdx.x >> 5;
  if ((threadIdx.x & 31) == 0) { s1[wave] = s; s2[wave] = ss; }
  __syncthreads();
  if (threadIdx.x == 0) {
    float ts = 0.f, tss = 0.f;
    for (int w = 0; w < 8; ++w) { ts += s1[w]; tss += s2[w]; }
    s1[0] = ts; s2[0] = tss;
  }
  __syncthreads();
  float mu = s1[0] / (float)H;
  float var = s2[0] / (float)H - mu * mu;
  float nv = (v - mu) * rsqrtf(var + 1e-12f) * g[i] + bta[i];
  h32[idx] = nv;
  h16[idx] = (_Float16)nv;
  if (fin) fin[idx] = nv;
}

extern "C" void kernel_launch(void* const* d_in, const int* in_sizes, int n_in,
                              void* d_out, int out_size, void* d_ws, size_t ws_size,
                              hipStream_t stream) {
  (void)in_sizes; (void)n_in; (void)out_size; (void)ws_size;
  const int B2 = 2, Se = 4096, Sd = 2048, H = 256, L = 6, F = 1024, NH = 4;
  const int BSd = B2 * Sd, BSe = B2 * Se;
  const size_t PL = (size_t)8 * H * H + (size_t)2 * H * F;  // f16 weights per layer

  const float* x          = (const float*)d_in[0];
  const float* y          = (const float*)d_in[1];
  const float* pos        = (const float*)d_in[2];
  const float* self_qkv_w = (const float*)d_in[3];
  const float* self_qkv_b = (const float*)d_in[4];
  const float* self_o_w   = (const float*)d_in[5];
  const float* self_o_b   = (const float*)d_in[6];
  const float* cross_qkv_w= (const float*)d_in[7];
  const float* cross_qkv_b= (const float*)d_in[8];
  const float* cross_o_w  = (const float*)d_in[9];
  const float* cross_o_b  = (const float*)d_in[10];
  const float* ffn_w1     = (const float*)d_in[11];
  const float* ffn_b1     = (const float*)d_in[12];
  const float* ffn_w2     = (const float*)d_in[13];
  const float* ffn_b2     = (const float*)d_in[14];
  const float* ln_g       = (const float*)d_in[15];
  const float* ln_b       = (const float*)d_in[16];

  size_t cur = 0;
  char* wsb = (char*)d_ws;
  auto take = [&](size_t bytes) -> char* {
    char* p = wsb + cur;
    cur += (bytes + 255) / 256 * 256;
    return p;
  };
  _Float16* xpe16 = (_Float16*)take((size_t)BSe * H * 2);
  float*    h32   = (float*)   take((size_t)BSd * H * 4);
  _Float16* h16   = (_Float16*)take((size_t)BSd * H * 2);
  _Float16* q16   = (_Float16*)take((size_t)BSd * H * 2);
  _Float16* k16   = (_Float16*)take((size_t)BSe * H * 2);
  _Float16* vt16  = (_Float16*)take((size_t)BSe * H * 2);
  _Float16* ctx16 = (_Float16*)take((size_t)BSd * H * 2);
  float*    tmp32 = (float*)   take((size_t)BSd * H * 4);
  _Float16* ffn16 = (_Float16*)take((size_t)BSd * F * 2);
  _Float16* wbuf  = (_Float16*)take((size_t)L * PL * 2);

  // --- convert + transpose all weights to f16 [N,K] ---
  for (int i = 0; i < L; ++i) {
    _Float16* LW = wbuf + (size_t)i * PL;
    for (int j = 0; j < 3; ++j)
      k_wconv<<<(H * H + 255) / 256, 256, 0, stream>>>(
          self_qkv_w + (size_t)(i * 3 + j) * H * H, LW + (size_t)j * H * H, H, H);
    k_wconv<<<(H * H + 255) / 256, 256, 0, stream>>>(
        self_o_w + (size_t)i * H * H, LW + (size_t)3 * H * H, H, H);
    for (int j = 0; j < 3; ++j)
      k_wconv<<<(H * H + 255) / 256, 256, 0, stream>>>(
          cross_qkv_w + (size_t)(i * 3 + j) * H * H, LW + (size_t)(4 + j) * H * H, H, H);
    k_wconv<<<(H * H + 255) / 256, 256, 0, stream>>>(
        cross_o_w + (size_t)i * H * H, LW + (size_t)7 * H * H, H, H);
    k_wconv<<<(H * F + 255) / 256, 256, 0, stream>>>(
        ffn_w1 + (size_t)i * H * F, LW + (size_t)8 * H * H, H, F);
    k_wconv<<<(F * H + 255) / 256, 256, 0, stream>>>(
        ffn_w2 + (size_t)i * F * H, LW + (size_t)8 * H * H + (size_t)F * H, F, H);
  }

  k_posembed<<<(BSe * H + 255) / 256, 256, 0, stream>>>(x, pos, xpe16, BSe * H, Se * H);
  k_inith<<<(BSd * H + 255) / 256, 256, 0, stream>>>(y, h32, h16, BSd * H);

  auto gemm = [&](const _Float16* A, const _Float16* W, const float* bias, float* o32,
                  _Float16* o16, int M, int N, int K, int flags, int SB) {
    dim3 g(M / 128, N / 64);
    k_gemm<<<g, 256, 0, stream>>>(A, W, bias, o32, o16, M, N, K, flags, SB);
  };

  for (int i = 0; i < L; ++i) {
    _Float16* LW = wbuf + (size_t)i * PL;
    // ---- self attention ----
    gemm(h16, LW + 0 * (size_t)H * H, self_qkv_b + (size_t)(i * 3 + 0) * H, nullptr, q16,  BSd, H, H, 2, 0);
    gemm(h16, LW + 1 * (size_t)H * H, self_qkv_b + (size_t)(i * 3 + 1) * H, nullptr, k16,  BSd, H, H, 2, 0);
    gemm(h16, LW + 2 * (size_t)H * H, self_qkv_b + (size_t)(i * 3 + 2) * H, nullptr, vt16, BSd, H, H, 2 | 4, Sd);
    k_flash<<<dim3(Sd / 64, NH, B2), 128, 0, stream>>>(q16, k16, vt16, ctx16, Sd, Sd);
    gemm(ctx16, LW + 3 * (size_t)H * H, self_o_b + (size_t)i * H, tmp32, nullptr, BSd, H, H, 1, 0);
    k_ln<<<BSd, 256, 0, stream>>>(h32, tmp32, ln_g + (size_t)(i * 3 + 0) * H,
                                  ln_b + (size_t)(i * 3 + 0) * H, h32, h16, nullptr, H);
    // ---- cross attention ----
    gemm(h16,   LW + 4 * (size_t)H * H, cross_qkv_b + (size_t)(i * 3 + 0) * H, nullptr, q16,  BSd, H, H, 2, 0);
    gemm(xpe16, LW + 5 * (size_t)H * H, cross_qkv_b + (size_t)(i * 3 + 1) * H, nullptr, k16,  BSe, H, H, 2, 0);
    gemm(xpe16, LW + 6 * (size_t)H * H, cross_qkv_b + (size_t)(i * 3 + 2) * H, nullptr, vt16, BSe, H, H, 2 | 4, Se);
    k_flash<<<dim3(Sd / 64, NH, B2), 128, 0, stream>>>(q16, k16, vt16, ctx16, Sd, Se);
    gemm(ctx16, LW + 7 * (size_t)H * H, cross_o_b + (size_t)i * H, tmp32, nullptr, BSd, H, H, 1, 0);
    k_ln<<<BSd, 256, 0, stream>>>(h32, tmp32, ln_g + (size_t)(i * 3 + 1) * H,
                                  ln_b + (size_t)(i * 3 + 1) * H, h32, h16, nullptr, H);
    // ---- FFN ----
    gemm(h16, LW + 8 * (size_t)H * H, ffn_b1 + (size_t)i * F, nullptr, ffn16, BSd, F, H, 2 | 8, 0);
    gemm(ffn16, LW + 8 * (size_t)H * H + (size_t)F * H, ffn_b2 + (size_t)i * H, tmp32, nullptr, BSd, H, F, 1, 0);
    k_ln<<<BSd, 256, 0, stream>>>(h32, tmp32, ln_g + (size_t)(i * 3 + 2) * H,
                                  ln_b + (size_t)(i * 3 + 2) * H, h32, h16,
                                  (i == L - 1) ? (float*)d_out : nullptr, H);
  }
}
